// Graphs_21500606284176
// MI455X (gfx1250) — matrix operand order repaired and checked by reading, not compiled, not run on hardware
//
#include <hip/hip_runtime.h>

typedef __attribute__((ext_vector_type(16))) _Float16 v16h;
typedef __attribute__((ext_vector_type(8)))  float    v8f;

namespace {
constexpr int G_ = 4, B_ = 128, N_ = 255, L_ = 256, F_ = 512;
constexpr int NL = N_ + L_;   // 511 total rows (internal + leaves)
constexpr int NP = 512;       // padded output-node rows
constexpr int KP = 512;       // padded contraction: [M0(256) | dM(256)]
}

// A-fragment layout for V_WMMA_F32_16X16X32_F16 (16x32 f16 A tile),
// contiguous k order: lane = (M&15)+16*((K>>4)&1), h = K&15.
// AM storage: [g][kt(16)][nt(32)][lane(32)][h(16)] so each lane's fragment
// is one contiguous 32-byte chunk.
__device__ __forceinline__ size_t am_index(int g, int n, int k) {
  int kt = k >> 5, nt = n >> 4;
  int lane = (n & 15) + 16 * ((k >> 4) & 1);
  int h    = k & 15;
  return ((((size_t)((g * 16 + kt) * 32 + nt)) * 32 + lane) * 16 + h);
}

// ---- CDNA5 async-to-LDS helpers (ASYNCcnt-tracked DMA, ISA ch.10/15.18) ----
// Copies one lane's 32-byte A-fragment slice into LDS. INST_OFFSET applies to
// both the memory and LDS addresses, so offset:16 covers the second half.
__device__ __forceinline__ void async_copy_frag(unsigned lds_off,
                                                const _Float16* gsrc) {
  asm volatile(
      "global_load_async_to_lds_b128 %0, %1, off\n\t"
      "global_load_async_to_lds_b128 %0, %1, off offset:16"
      :: "v"(lds_off), "v"((unsigned long long)(uintptr_t)gsrc)
      : "memory");
}
__device__ __forceinline__ void wait_async_le2() {
  asm volatile("s_wait_asynccnt 2" ::: "memory");
}
__device__ __forceinline__ void wait_async_le0() {
  asm volatile("s_wait_asynccnt 0" ::: "memory");
}

__global__ void k_zero_h(_Float16* p, int n) {
  int t = blockIdx.x * blockDim.x + threadIdx.x;
  if (t < n) p[t] = (_Float16)0.0f;
}

// x (B,F) f32 -> A-fragment tiles xA[bt(8)][kt(16)][lane][h] (M=batch, K=f)
__global__ void k_prep_x(const float* x, _Float16* xA) {
  int wid  = (blockIdx.x * blockDim.x + threadIdx.x) >> 5;
  int lane = threadIdx.x & 31;
  int bt = wid >> 4, kt = wid & 15;
  int b = bt * 16 + (lane & 15);
  int s = (lane >> 4) & 1;
  _Float16* dst = xA + (((size_t)(bt * 16 + kt) * 32 + lane) * 16);
#pragma unroll
  for (int h = 0; h < 16; ++h) {
    int K = h + 16 * s;
    dst[h] = (_Float16)x[(size_t)b * F_ + kt * 32 + K];
  }
}

// W (G,N,F) f32 -> B-fragment tiles WB[g][kt(16)][nt(16)][lane][h]
// B layout (32x16, K x node): lane = (node&15)+16*(Klocal>>4), h = Klocal&15.
__global__ void k_prep_w(const float* W, _Float16* WB) {
  int wid  = (blockIdx.x * blockDim.x + threadIdx.x) >> 5;
  int lane = threadIdx.x & 31;
  int g = wid >> 8;
  int kt = (wid >> 4) & 15;
  int nt = wid & 15;
  int node = nt * 16 + (lane & 15);
  int fbase = kt * 32 + 16 * (lane >> 4);
  _Float16* dst = WB + ((((size_t)(g * 16 + kt) * 16 + nt) * 32 + lane) * 16);
#pragma unroll
  for (int h = 0; h < 16; ++h) {
    float v = (node < N_) ? W[((size_t)g * N_ + node) * F_ + fbase + h] : 0.0f;
    dst[h] = (_Float16)v;
  }
}

// Column softmax (axis=1, 511 rows) of M_left / M_right for one (g,m) per
// wave; writes M0 into AM[:, k=m] and dM=M1-M0 into AM[:, k=256+m] (f16,
// A-fragment layout). exp without max-shift is exact softmax and safe here:
// inputs are uniform[0,1].
__global__ void k_softmax(const float* Ml, const float* Mr, _Float16* AM) {
  int wid  = (blockIdx.x * blockDim.x + threadIdx.x) >> 5;
  int lane = threadIdx.x & 31;
  if (wid >= G_ * N_) return;
  int g = wid / N_, m = wid % N_;
  float eL[16], eR[16];
  float sL = 0.f, sR = 0.f;
#pragma unroll
  for (int i = 0; i < 16; ++i) {
    int r = lane + 32 * i;
    float a = 0.f, bv = 0.f;
    if (r < NL) {
      a  = expf(Ml[((size_t)g * NL + r) * N_ + m]);
      bv = expf(Mr[((size_t)g * NL + r) * N_ + m]);
    }
    eL[i] = a; eR[i] = bv; sL += a; sR += bv;
  }
#pragma unroll
  for (int off = 16; off > 0; off >>= 1) {
    sL += __shfl_xor(sL, off, 32);
    sR += __shfl_xor(sR, off, 32);
  }
  float invL = 1.f / sL, invR = 1.f / sR;
#pragma unroll
  for (int i = 0; i < 16; ++i) {
    int r = lane + 32 * i;
    if (r < NL) {
      float m0 = eL[i] * invL;
      float dm = eR[i] * invR - m0;
      AM[am_index(g, r, m)]       = (_Float16)m0;
      AM[am_index(g, r, m + 256)] = (_Float16)dm;
    }
  }
}

// logits[g,b,n] = x@W^T via WMMA, then bprob = sigmoid(logits + bias)
__global__ void k_logits(const _Float16* xA, const _Float16* WB,
                         const float* bias, float* bprob) {
  int wid  = (blockIdx.x * blockDim.x + threadIdx.x) >> 5;
  int lane = threadIdx.x & 31;
  int g  = wid >> 7;
  int bt = (wid >> 4) & 7;
  int nt = wid & 15;
  v8f acc = {};
#pragma unroll 4
  for (int kt = 0; kt < 16; ++kt) {
    v16h a = *reinterpret_cast<const v16h*>(
        xA + (((size_t)(bt * 16 + kt) * 32 + lane) * 16));
    v16h b = *reinterpret_cast<const v16h*>(
        WB + ((((size_t)(g * 16 + kt) * 16 + nt) * 32 + lane) * 16));
    acc = __builtin_amdgcn_wmma_f32_16x16x32_f16(false, a, false, b,
                                                 (short)0, acc, false, false);
  }
  int node = nt * 16 + (lane & 15);
  int brow = bt * 16 + 8 * (lane >> 4);
#pragma unroll
  for (int r = 0; r < 8; ++r) {
    float v = 0.f;
    if (node < N_) {
      float z = acc[r] + bias[g * N_ + node];
      v = 1.f / (1.f + expf(-z));
    }
    bprob[((size_t)g * B_ + (brow + r)) * 256 + node] = v;
  }
}

__global__ void k_init_prob(float* prob) {
  int t = blockIdx.x * blockDim.x + threadIdx.x;
  prob[t] = ((t & (NP - 1)) == 0) ? 1.f : 0.f;  // unit mass on root node 0
}

// One propagation step, fused: block = (g, batch-tile). The stacked operand
// [s;t] (16 batch cols x 512 k, f16) is built straight into LDS and reused by
// all 32 node-tiles; each wave computes 4 node-tiles, streaming its A
// fragments through a per-wave LDS double buffer via async-to-LDS DMA.
__global__ void __launch_bounds__(256) k_prop(const _Float16* AM,
                                              const float* bprob,
                                              const float* probCur,
                                              float* probNext) {
  __shared__ _Float16 stLDS[16 * KP];      // 16 KiB: [bl(16)][k(512)]
  __shared__ _Float16 aLDS[8][2][512];     // 16 KiB: per-wave double buffer
  int g  = blockIdx.x >> 3;
  int bt = blockIdx.x & 7;
  int tid  = threadIdx.x;
  int wave = tid >> 5, lane = tid & 31;

  // ---- build [s;t] tile in LDS (replaces a separate kernel + global trip) --
  {
    int bl = tid >> 4;                 // 0..15 local batch col
    int kbase = (tid & 15) * 32;       // 32 contiguous k per thread
    size_t gb = (size_t)g * B_ + bt * 16 + bl;
#pragma unroll
    for (int i = 0; i < 32; ++i) {
      int k = kbase + i;
      float v = 0.f;
      if (k < N_) {
        v = probCur[gb * NP + k];
      } else if (k >= 256) {
        int m = k - 256;
        if (m < N_) v = bprob[gb * 256 + m] * probCur[gb * NP + m];
      }
      stLDS[bl * KP + k] = (_Float16)v;
    }
  }
  __syncthreads();

  const _Float16* Ag = AM + (size_t)g * (16 * 32 * 32 * 16);
  unsigned aoff0 = (unsigned)(uintptr_t)(&aLDS[wave][0][lane * 16]);
  unsigned aoff1 = (unsigned)(uintptr_t)(&aLDS[wave][1][lane * 16]);
  int boff = (lane & 15) * KP + 16 * (lane >> 4);
  int bcol = bt * 16 + (lane & 15);

  for (int ntl = 0; ntl < 4; ++ntl) {
    int nt = wave * 4 + ntl;
    // prime tile kt=0 into buffer 0
    async_copy_frag(aoff0, Ag + (((size_t)(0 * 32 + nt)) * 32 + lane) * 16);
    v8f acc = {};
#pragma unroll
    for (int kt = 0; kt < 16; ++kt) {
      if (kt < 15)  // prefetch next tile into the other buffer
        async_copy_frag((kt & 1) ? aoff0 : aoff1,
                        Ag + (((size_t)((kt + 1) * 32 + nt)) * 32 + lane) * 16);
      if (kt < 15) wait_async_le2(); else wait_async_le0();
      v16h a = *reinterpret_cast<const v16h*>(&aLDS[wave][kt & 1][lane * 16]);
      v16h b = *reinterpret_cast<const v16h*>(&stLDS[boff + kt * 32]);
      acc = __builtin_amdgcn_wmma_f32_16x16x32_f16(false, a, false, b,
                                                   (short)0, acc, false, false);
    }
    int n0 = nt * 16 + 8 * (lane >> 4);
    size_t base = ((size_t)g * B_ + bcol) * NP + n0;
#pragma unroll
    for (int r = 0; r < 8; ++r) {
      int n = n0 + r;
      float keep = (n >= N_ && n < NL) ? probCur[base + r] : 0.f;  // absorbing
      probNext[base + r] = acc[r] + keep;
    }
  }
}

// out (B,L,G) = interval(prob[..., N:]) transposed
__global__ void k_finalize(const float* prob, float* out) {
  int t = blockIdx.x * blockDim.x + threadIdx.x;  // (b*L + l)*G + g
  int g = t & 3;
  int l = (t >> 2) & (L_ - 1);
  int b = t >> 10;
  float v = prob[((size_t)g * B_ + b) * NP + N_ + l];
  v = (v > 0.f) ? v : 1e-5f;
  v = (v < 1.f) ? v : (1.f - 1e-5f);
  out[t] = v;
}

extern "C" void kernel_launch(void* const* d_in, const int* in_sizes, int n_in,
                              void* d_out, int out_size, void* d_ws, size_t ws_size,
                              hipStream_t stream) {
  (void)in_sizes; (void)n_in; (void)out_size; (void)ws_size;
  const float* x    = (const float*)d_in[0];
  const float* W    = (const float*)d_in[1];
  const float* bias = (const float*)d_in[2];
  const float* Ml   = (const float*)d_in[3];
  const float* Mr   = (const float*)d_in[4];
  float* out = (float*)d_out;

  char* ws = (char*)d_ws;
  _Float16* AM    = (_Float16*)(ws);                                  // 2 MiB
  _Float16* WB    = (_Float16*)(ws + (2u << 20));                     // 1 MiB
  _Float16* xA    = (_Float16*)(ws + (3u << 20));                     // 128 KiB
  float*    bprob = (float*)   (ws + (3u << 20) + (128u << 10));      // 512 KiB
  float*    pA    = (float*)   (ws + (3u << 20) + (640u << 10));      // 1 MiB
  float*    pB    = (float*)   (ws + (4u << 20) + (640u << 10));      // 1 MiB

  dim3 blk(256);
  k_zero_h   <<<4096, blk, 0, stream>>>(AM, 4 * 16 * 32 * 32 * 16);
  k_prep_x   <<<16,   blk, 0, stream>>>(x, xA);
  k_prep_w   <<<128,  blk, 0, stream>>>(W, WB);
  k_softmax  <<<128,  blk, 0, stream>>>(Ml, Mr, AM);
  k_logits   <<<64,   blk, 0, stream>>>(xA, WB, bias, bprob);
  k_init_prob<<<1024, blk, 0, stream>>>(pA);

  float* cur = pA; float* nxt = pB;
  for (int j = 0; j < 8; ++j) {
    k_prop<<<32, blk, 0, stream>>>(AM, bprob, cur, nxt);
    float* tmp = cur; cur = nxt; nxt = tmp;
  }
  k_finalize<<<512, blk, 0, stream>>>(cur, out);
}
